// Axial_Layer_17978733101092
// MI455X (gfx1250) — matrix operand order repaired and checked by reading, not compiled, not run on hardware
//
#include <hip/hip_runtime.h>
#include <hip/hip_bf16.h>

typedef __attribute__((ext_vector_type(16))) __bf16 v16bf;
typedef __attribute__((ext_vector_type(8)))  __bf16 v8bf;
typedef __attribute__((ext_vector_type(4)))  __bf16 v4bf;
typedef __attribute__((ext_vector_type(2)))  __bf16 v2bf;
typedef __attribute__((ext_vector_type(8)))  float  v8f;
typedef __attribute__((ext_vector_type(4)))  float  v4f;

#define NB    2
#define CIN   256
#define OC    512
#define SP    128     // spatial H = W = sequence length
#define NH    8
#define DHEAD 32
#define DQK   16
#define EPSV  1e-5f

// workspace byte offsets
#define OFF_KQV   0ull                 // bf16 [b][hsp][o][w]  33,554,432 B
#define OFF_QKVT  33554432ull          // bf16 [bw][o][x]      33,554,432 B
#define OFF_REL   67108864ull          // f32  [64][128][128]   4,194,304 B
#define OFF_ATTN  71303168ull          // f32  [bw][c=256][x]  33,554,432 B
#define OFF_COEF  104857600ull         // f32 scale[512], shift[512]
#define OFF_LSUM  (OFF_COEF + 4096ull) // f32 [24]
#define OFF_LSQ   (OFF_LSUM + 96ull)   // f32 [24]
#define OFF_LA    (OFF_LSQ + 96ull)    // f32 [24]
#define OFF_LC    (OFF_LA + 96ull)     // f32 [24]

__device__ inline v8f v8f_zero() {
  v8f v;
  #pragma unroll
  for (int i = 0; i < 8; ++i) v[i] = 0.f;
  return v;
}
__device__ inline v4f v4f_zero() {
  v4f v;
  #pragma unroll
  for (int i = 0; i < 4; ++i) v[i] = 0.f;
  return v;
}
__device__ inline v8bf v8bf_zero() {
  v8bf v;
  #pragma unroll
  for (int i = 0; i < 8; ++i) v[i] = (__bf16)0.f;
  return v;
}
__device__ inline v16bf cat8(v8bf lo, v8bf hi) {
  return __builtin_shufflevector(lo, hi, 0, 1, 2, 3, 4, 5, 6, 7,
                                 8, 9, 10, 11, 12, 13, 14, 15);
}
__device__ inline v8f wmma_bf16(v16bf a, v16bf b, v8f c) {
  // D(16x16 f32) = A(16x32 bf16) x B(32x16 bf16) + C
  return __builtin_amdgcn_wmma_f32_16x16x32_bf16(false, a, false, b, (short)0, c,
                                                 false, false);
}
__device__ inline float wave_sum(float v) {
  #pragma unroll
  for (int o = 16; o; o >>= 1) v += __shfl_xor(v, o, 32);
  return v;
}

// ---------------------------------------------------------------------------
// K1: kqv GEMM: per (b,hsp) batch, out[o][w] = sum_c W[o][c] * x[b][c][hsp][w]
// A kept [M][K] (frag = 2x contiguous 16B); B stored transposed [N][K].
// Staging fully packed: register-batched loads, b128 LDS stores.
// ---------------------------------------------------------------------------
__global__ void __launch_bounds__(256)
k_kqv_gemm(const float* __restrict__ x, const float* __restrict__ wk,
           __bf16* __restrict__ kqv) {
  __shared__ __attribute__((aligned(16))) __bf16 As[64][32];
  __shared__ __attribute__((aligned(16))) __bf16 BsT[128][40];
  __shared__ __attribute__((aligned(16))) __bf16 Cs[64][128];
  const int bat = blockIdx.x;           // b*128 + hsp
  const int b = bat >> 7, hsp = bat & 127;
  const int o0 = blockIdx.y * 64;
  const int tid = threadIdx.x;
  const int wid = tid >> 5, lane = tid & 31;
  const int half = lane >> 4, l15 = lane & 15;
  const int mbase = (wid >> 1) * 16;
  const int nbase = (wid & 1) * 64;
  const int ar = tid >> 2, acg = (tid & 3) * 8;   // A staging map
  const int bw_ = tid & 127;                      // B staging map
  v8f acc[4];
  #pragma unroll
  for (int t = 0; t < 4; ++t) acc[t] = v8f_zero();

  for (int kk = 0; kk < CIN; kk += 32) {
    {                                   // A chunk 64x32: 8 contiguous floats/thread
      float av[8];
      #pragma unroll
      for (int c = 0; c < 8; ++c)
        av[c] = wk[(o0 + ar) * CIN + kk + acg + c];
      v8bf pk;
      #pragma unroll
      for (int c = 0; c < 8; ++c) pk[c] = (__bf16)av[c];
      *(v8bf*)&As[ar][acg] = pk;
    }
    #pragma unroll
    for (int jr = 0; jr < 2; ++jr) {    // B chunk 32x128: 8 K-rows/thread, packed
      int rg = (tid >> 7) + jr * 2;     // group of 8 rows
      float xv[8];
      #pragma unroll
      for (int r = 0; r < 8; ++r)
        xv[r] = x[(((size_t)b * CIN + kk + rg * 8 + r) * SP + hsp) * SP + bw_];
      v8bf pk;
      #pragma unroll
      for (int r = 0; r < 8; ++r) pk[r] = (__bf16)xv[r];
      *(v8bf*)&BsT[bw_][rg * 8] = pk;
    }
    if (kk + 32 < CIN)                  // prefetch next x chunk into GL2
      __builtin_prefetch(&x[(((size_t)b * CIN + kk + 32 + (tid >> 7)) * SP + hsp) * SP +
                            (tid & 127)], 0, 3);
    __syncthreads();
    const __bf16* arow = &As[mbase + l15][0];
    v16bf a = cat8(*(const v8bf*)(arow + half * 16),
                   *(const v8bf*)(arow + half * 16 + 8));
    #pragma unroll
    for (int t = 0; t < 4; ++t) {
      const __bf16* brow = &BsT[nbase + t * 16 + l15][half * 16];
      v16bf bb = cat8(*(const v8bf*)brow, *(const v8bf*)(brow + 8));
      acc[t] = wmma_bf16(a, bb, acc[t]);
    }
    __syncthreads();
  }
  // C/D layout: VGPR r -> M=r (lanes 0-15) / M=r+8 (lanes 16-31), N=lane&15
  #pragma unroll
  for (int t = 0; t < 4; ++t)
    #pragma unroll
    for (int r = 0; r < 8; ++r)
      Cs[mbase + r + (half << 3)][nbase + t * 16 + l15] = (__bf16)acc[t][r];
  __syncthreads();
  #pragma unroll
  for (int j = 0; j < 4; ++j) {         // contiguous 16B global stores
    int i = tid + j * 256, ol = i >> 4, wc = (i & 15) * 8;
    *(v8bf*)&kqv[(((size_t)bat) * OC + o0 + ol) * SP + wc] =
        *(const v8bf*)&Cs[ol][wc];
  }
}

// ---------------------------------------------------------------------------
// K2: BN stats for kqv -> per-channel affine (scale, shift)
// ---------------------------------------------------------------------------
__global__ void __launch_bounds__(256)
k_kqv_stats(const __bf16* __restrict__ kqv, const float* __restrict__ gamma,
            const float* __restrict__ beta, float* __restrict__ scale,
            float* __restrict__ shift) {
  __shared__ float rs[256], rq[256];
  const int o = blockIdx.x, tid = threadIdx.x;
  float s = 0.f, q = 0.f;
  #pragma unroll 4
  for (int i = tid; i < 4096; i += 256) {   // 256 rows x 16 chunks of 8
    int r = i >> 4, wc = (i & 15) * 8;
    v8bf v = *(const v8bf*)&kqv[(((size_t)r) * OC + o) * SP + wc];
    #pragma unroll
    for (int u = 0; u < 8; ++u) { float f = (float)v[u]; s += f; q += f * f; }
  }
  rs[tid] = s; rq[tid] = q;
  __syncthreads();
  for (int st = 128; st > 0; st >>= 1) {
    if (tid < st) { rs[tid] += rs[tid + st]; rq[tid] += rq[tid + st]; }
    __syncthreads();
  }
  if (tid == 0) {
    float n = 256.f * SP;
    float mean = rs[0] / n;
    float var = rq[0] / n - mean * mean;
    float sc = gamma[o] * rsqrtf(var + EPSV);
    scale[o] = sc;
    shift[o] = beta[o] - mean * sc;
  }
}

// ---------------------------------------------------------------------------
// K3: normalize + transpose (hsp <-> w) into attention layout [bw][o][x] bf16
// ---------------------------------------------------------------------------
__global__ void __launch_bounds__(256)
k_transpose_norm(const __bf16* __restrict__ kqv, const float* __restrict__ scale,
                 const float* __restrict__ shift, __bf16* __restrict__ qkvt) {
  __shared__ float tile[32][33];
  const int bo = blockIdx.x;            // b*512 + o
  const int b = bo >> 9, o = bo & 511;
  const int hsp0 = (blockIdx.y >> 2) * 32, w0 = (blockIdx.y & 3) * 32;
  const int col = threadIdx.x & 31, rg = threadIdx.x >> 5;
  const float sc = scale[o], sh = shift[o];
  #pragma unroll
  for (int rr = 0; rr < 4; ++rr) {
    int row = rg * 4 + rr;
    float v = (float)kqv[(((size_t)(b * 128 + hsp0 + row)) * OC + o) * SP + w0 + col];
    tile[row][col] = v * sc + sh;
  }
  __syncthreads();
  #pragma unroll
  for (int rr = 0; rr < 4; ++rr) {
    int row = rg * 4 + rr;              // w_local
    qkvt[(((size_t)(b * 128 + w0 + row)) * OC + o) * SP + hsp0 + col] =
        (__bf16)tile[col][row];
  }
}

// ---------------------------------------------------------------------------
// K4: gather rel[d][x][y] = rel_encoding[d][y - x + 127]
// ---------------------------------------------------------------------------
__global__ void k_rel_gather(const float* __restrict__ re, float* __restrict__ rel) {
  int d = blockIdx.x, xx = blockIdx.y, y = threadIdx.x;
  rel[((size_t)d * SP + xx) * SP + y] = re[d * 255 + (y - xx + 127)];
}

__global__ void k_zero(float* p, int n) {
  int i = blockIdx.x * blockDim.x + threadIdx.x;
  if (i < n) p[i] = 0.f;
}

// ---------------------------------------------------------------------------
// staging helper: load q,k (bf16, rows contiguous in x) into transposed
// [x][d] LDS tiles with d=16..31 zero-filled (uniform frag vector loads)
// ---------------------------------------------------------------------------
__device__ inline void stage_qk(const __bf16* __restrict__ qkvt, size_t base,
                                __bf16 (*qsT)[40], __bf16 (*ksT)[40], int tid) {
  {
    int zr = tid >> 1, zc = 16 + (tid & 1) * 8;
    v8bf z = v8bf_zero();
    *(v8bf*)&qsT[zr][zc] = z;
    *(v8bf*)&ksT[zr][zc] = z;
  }
  int d0 = (tid >> 5) * 2, xc = (tid & 31) * 4;   // d-pair, 4 x's per thread
  v4bf k0 = *(const v4bf*)&qkvt[base + (size_t)d0 * SP + xc];
  v4bf k1 = *(const v4bf*)&qkvt[base + (size_t)(d0 + 1) * SP + xc];
  v4bf q0 = *(const v4bf*)&qkvt[base + (size_t)(16 + d0) * SP + xc];
  v4bf q1 = *(const v4bf*)&qkvt[base + (size_t)(17 + d0) * SP + xc];
  #pragma unroll
  for (int u = 0; u < 4; ++u) {
    v2bf kp; kp[0] = k0[u]; kp[1] = k1[u];
    v2bf qp; qp[0] = q0[u]; qp[1] = q1[u];
    *(v2bf*)&ksT[xc + u][d0] = kp;
    *(v2bf*)&qsT[xc + u][d0] = qp;
  }
}

// ---------------------------------------------------------------------------
// K5: logits stats pass (qk via WMMA, qr/kr via vectorized VALU)
// ---------------------------------------------------------------------------
__global__ void __launch_bounds__(256)
k_logit_stats(const __bf16* __restrict__ qkvt, const float* __restrict__ rel,
              float* __restrict__ lsum, float* __restrict__ lsq) {
  __shared__ __attribute__((aligned(16))) __bf16 qsT[128][40];
  __shared__ __attribute__((aligned(16))) __bf16 ksT[128][40];
  __shared__ float red[8][6];
  const int bw = blockIdx.x, h = blockIdx.y;
  const int tid = threadIdx.x, wid = tid >> 5, lane = tid & 31;
  const int half = lane >> 4, l15 = lane & 15;
  const size_t base = ((size_t)bw * OC + h * 64) * SP;
  stage_qk(qkvt, base, qsT, ksT, tid);
  __syncthreads();
  const int xb = wid * 16;
  const __bf16* qrow = &qsT[xb + l15][0];
  v16bf a = cat8(*(const v8bf*)(qrow + half * 16),
                 *(const v8bf*)(qrow + half * 16 + 8));   // A = q^T, K padded 0
  float s0 = 0, q0 = 0, s1 = 0, q1 = 0, s2 = 0, q2 = 0;
  #pragma unroll
  for (int yt = 0; yt < 8; ++yt) {
    const __bf16* krow = &ksT[yt * 16 + l15][half * 16];
    v16bf bb = cat8(*(const v8bf*)krow, *(const v8bf*)(krow + 8));
    v8f c = wmma_bf16(a, bb, v8f_zero());
    #pragma unroll
    for (int r = 0; r < 8; ++r) { float v = c[r]; s0 += v; q0 += v * v; }
  }
  // qr: contiguous in y -> v4f loads; lane covers (xl, 4 y's)
  for (int i = lane; i < 512; i += 32) {
    int xl = i >> 5, y4 = (i & 31) * 4, xg = xb + xl;
    v4f aq = v4f_zero();
    #pragma unroll
    for (int d = 0; d < DQK; ++d) {
      float qd = (float)qsT[xg][d];
      v4f rv = *(const v4f*)&rel[((size_t)d * SP + xg) * SP + y4];
      aq += qd * rv;
    }
    #pragma unroll
    for (int j = 0; j < 4; ++j) { s1 += aq[j]; q1 += aq[j] * aq[j]; }
  }
  // kr: contiguous in x -> v4f loads; lane covers (y, 4 x's)
  for (int i = lane; i < 512; i += 32) {
    int y = i >> 2, x4 = (i & 3) * 4;
    v4f ak = v4f_zero();
    #pragma unroll
    for (int d = 0; d < DQK; ++d) {
      float kd = (float)ksT[y][d];
      v4f rv = *(const v4f*)&rel[((size_t)(16 + d) * SP + y) * SP + xb + x4];
      ak += kd * rv;
    }
    #pragma unroll
    for (int j = 0; j < 4; ++j) { s2 += ak[j]; q2 += ak[j] * ak[j]; }
  }
  s0 = wave_sum(s0); q0 = wave_sum(q0); s1 = wave_sum(s1);
  q1 = wave_sum(q1); s2 = wave_sum(s2); q2 = wave_sum(q2);
  if (lane == 0) {
    red[wid][0] = s0; red[wid][1] = q0; red[wid][2] = s1;
    red[wid][3] = q1; red[wid][4] = s2; red[wid][5] = q2;
  }
  __syncthreads();
  if (tid < 6) {
    float t = 0.f;
    for (int w = 0; w < 8; ++w) t += red[w][tid];
    int ch = (tid >> 1) * NH + h;                      // part*8 + h
    if (tid & 1) atomicAdd(&lsq[ch], t); else atomicAdd(&lsum[ch], t);
  }
}

// ---------------------------------------------------------------------------
// K6: finalize BN-affine coefficients for the 24 logits channels
// ---------------------------------------------------------------------------
__global__ void k_logit_fin(const float* __restrict__ lsum, const float* __restrict__ lsq,
                            const float* __restrict__ gamma, const float* __restrict__ beta,
                            float* __restrict__ la, float* __restrict__ lc) {
  int c = threadIdx.x;
  if (c < 24) {
    float n = 256.f * SP * SP;
    float mean = lsum[c] / n;
    float var = lsq[c] / n - mean * mean;
    float aa = gamma[c] * rsqrtf(var + EPSV);
    la[c] = aa;
    lc[c] = beta[c] - mean * aa;
  }
}

// ---------------------------------------------------------------------------
// K7: fused logits(recompute) + softmax + attention per (bw,h)
// strip/wbuf/enc are wave-private ([wid]); DS ops from one wave are in-order,
// so no barriers are needed after the staging barrier.
// ---------------------------------------------------------------------------
__global__ void __launch_bounds__(256)
k_attn(const __bf16* __restrict__ qkvt, const float* __restrict__ rel,
       const float* __restrict__ la, const float* __restrict__ lc,
       float* __restrict__ attn_tmp) {
  __shared__ __attribute__((aligned(16))) __bf16 qsT[128][40];
  __shared__ __attribute__((aligned(16))) __bf16 ksT[128][40];
  __shared__ __attribute__((aligned(16))) __bf16 vs[DHEAD][136];
  __shared__ __attribute__((aligned(16))) __bf16 wbuf[8][16][136];
  __shared__ float strip[8][16][SP + 4];
  __shared__ float enc[8][DHEAD][16];
  const int bw = blockIdx.x, h = blockIdx.y;
  const int tid = threadIdx.x, wid = tid >> 5, lane = tid & 31;
  const int half = lane >> 4, l15 = lane & 15;
  const size_t base = ((size_t)bw * OC + h * 64) * SP;
  stage_qk(qkvt, base, qsT, ksT, tid);
  {
    int d = tid >> 4, xc = (tid & 15) * 8;     // v natural layout, b128<->b128
    *(v8bf*)&vs[d][xc]      = *(const v8bf*)&qkvt[base + (size_t)(32 + d) * SP + xc];
    *(v8bf*)&vs[16 + d][xc] = *(const v8bf*)&qkvt[base + (size_t)(48 + d) * SP + xc];
  }
  __syncthreads();
  const float a0 = la[h], a1 = la[8 + h], a2 = la[16 + h];
  const float cc = lc[h] + lc[8 + h] + lc[16 + h];
  const int xb = wid * 16;
  // qk via WMMA -> strip (affine-scaled)
  const __bf16* qrow = &qsT[xb + l15][0];
  v16bf a = cat8(*(const v8bf*)(qrow + half * 16),
                 *(const v8bf*)(qrow + half * 16 + 8));
  #pragma unroll
  for (int yt = 0; yt < 8; ++yt) {
    const __bf16* krow = &ksT[yt * 16 + l15][half * 16];
    v16bf bb = cat8(*(const v8bf*)krow, *(const v8bf*)(krow + 8));
    v8f c = wmma_bf16(a, bb, v8f_zero());
    #pragma unroll
    for (int r = 0; r < 8; ++r)
      strip[wid][r + (half << 3)][yt * 16 + l15] = c[r] * a0 + cc;
  }
  // qr: contiguous in y; vector RMW on strip rows
  for (int i = lane; i < 512; i += 32) {
    int xl = i >> 5, y4 = (i & 31) * 4, xg = xb + xl;
    v4f aq = v4f_zero();
    #pragma unroll
    for (int d = 0; d < DQK; ++d) {
      float qd = (float)qsT[xg][d];
      v4f rv = *(const v4f*)&rel[((size_t)d * SP + xg) * SP + y4];
      aq += qd * rv;
    }
    v4f cur = *(const v4f*)&strip[wid][xl][y4];
    cur += a1 * aq;
    *(v4f*)&strip[wid][xl][y4] = cur;
  }
  // kr: contiguous in x; scalar RMW on strip columns
  for (int i = lane; i < 512; i += 32) {
    int y = i >> 2, x4 = (i & 3) * 4;
    v4f ak = v4f_zero();
    #pragma unroll
    for (int d = 0; d < DQK; ++d) {
      float kd = (float)ksT[y][d];
      v4f rv = *(const v4f*)&rel[((size_t)(16 + d) * SP + y) * SP + xb + x4];
      ak += kd * rv;
    }
    #pragma unroll
    for (int j = 0; j < 4; ++j) strip[wid][x4 + j][y] += a2 * ak[j];
  }
  // softmax along y; normalized weights written as bf16 (frag-native layout)
  for (int rr = 0; rr < 16; ++rr) {
    float v0 = strip[wid][rr][lane],      v1 = strip[wid][rr][lane + 32];
    float v2 = strip[wid][rr][lane + 64], v3 = strip[wid][rr][lane + 96];
    float m = fmaxf(fmaxf(v0, v1), fmaxf(v2, v3));
    #pragma unroll
    for (int o = 16; o; o >>= 1) m = fmaxf(m, __shfl_xor(m, o, 32));
    float e0 = __expf(v0 - m), e1 = __expf(v1 - m);
    float e2 = __expf(v2 - m), e3 = __expf(v3 - m);
    float s = e0 + e1 + e2 + e3;
    #pragma unroll
    for (int o = 16; o; o >>= 1) s += __shfl_xor(s, o, 32);
    float inv = 1.f / s;
    wbuf[wid][rr][lane]      = (__bf16)(e0 * inv);
    wbuf[wid][rr][lane + 32] = (__bf16)(e1 * inv);
    wbuf[wid][rr][lane + 64] = (__bf16)(e2 * inv);
    wbuf[wid][rr][lane + 96] = (__bf16)(e3 * inv);
  }
  // attn_enc: out[d][xl] = sum_y w[xl][y] * v_enc[d][xb+xl][y]  (vectorized)
  for (int i = lane; i < 512; i += 32) {
    int xl = i >> 5, d = i & 31;
    v4f sv = v4f_zero();
    for (int y = 0; y < SP; y += 4) {
      v4bf wv = *(const v4bf*)&wbuf[wid][xl][y];
      v4f rv = *(const v4f*)&rel[((size_t)(32 + d) * SP + xb + xl) * SP + y];
      v4f wf;
      #pragma unroll
      for (int j = 0; j < 4; ++j) wf[j] = (float)wv[j];
      sv += wf * rv;
    }
    enc[wid][d][xl] = sv[0] + sv[1] + sv[2] + sv[3];
  }
  // attn via WMMA: D(x by d) = sum_y W(x by y) * v^T(y by d); K=128 in 4 chunks
  v8f o2[2];
  o2[0] = v8f_zero(); o2[1] = v8f_zero();
  #pragma unroll
  for (int kc = 0; kc < 4; ++kc) {
    const __bf16* wrow = &wbuf[wid][l15][kc * 32];
    v16bf aw = cat8(*(const v8bf*)(wrow + half * 16),
                    *(const v8bf*)(wrow + half * 16 + 8));
    #pragma unroll
    for (int n2 = 0; n2 < 2; ++n2) {
      const __bf16* vrow = &vs[n2 * 16 + l15][kc * 32 + half * 16];
      v16bf bv = cat8(*(const v8bf*)vrow, *(const v8bf*)(vrow + 8));
      o2[n2] = wmma_bf16(aw, bv, o2[n2]);
    }
  }
  #pragma unroll
  for (int n2 = 0; n2 < 2; ++n2)
    #pragma unroll
    for (int r = 0; r < 8; ++r)
      enc[wid][n2 * 16 + l15][r + (half << 3)] += o2[n2][r];
  // contiguous 16B stores into attn_tmp [bw][c=h*32+d][x]
  for (int i = lane; i < 128; i += 32) {
    int d = i >> 2, x4 = (i & 3) * 4;
    *(v4f*)&attn_tmp[(((size_t)bw) * 256 + h * DHEAD + d) * SP + xb + x4] =
        *(const v4f*)&enc[wid][d][x4];
  }
}

// ---------------------------------------------------------------------------
// K8: output transpose: out[b][c][x][w] = attn_tmp[(b*128+w)*256+c][x]
// ---------------------------------------------------------------------------
__global__ void __launch_bounds__(256)
k_out_transpose(const float* __restrict__ attn_tmp, float* __restrict__ out) {
  __shared__ float tile[32][33];
  const int bc = blockIdx.x;            // b*256 + c
  const int b = bc >> 8, c = bc & 255;
  const int w0 = (blockIdx.y >> 2) * 32, x0 = (blockIdx.y & 3) * 32;
  const int col = threadIdx.x & 31, rg = threadIdx.x >> 5;
  #pragma unroll
  for (int rr = 0; rr < 4; ++rr) {
    int row = rg * 4 + rr;              // w local
    tile[row][col] = attn_tmp[(((size_t)(b * 128 + w0 + row)) * 256 + c) * SP + x0 + col];
  }
  __syncthreads();
  #pragma unroll
  for (int rr = 0; rr < 4; ++rr) {
    int row = rg * 4 + rr;              // x local
    out[(((size_t)bc) * SP + x0 + row) * SP + w0 + col] = tile[col][row];
  }
}

extern "C" void kernel_launch(void* const* d_in, const int* in_sizes, int n_in,
                              void* d_out, int out_size, void* d_ws, size_t ws_size,
                              hipStream_t stream) {
  (void)in_sizes; (void)n_in; (void)out_size; (void)ws_size;
  const float* x  = (const float*)d_in[0];
  const float* wk = (const float*)d_in[1];
  const float* kg = (const float*)d_in[2];
  const float* kb = (const float*)d_in[3];
  const float* lg = (const float*)d_in[4];
  const float* lb = (const float*)d_in[5];
  const float* re = (const float*)d_in[6];
  char* ws = (char*)d_ws;
  __bf16* kqv   = (__bf16*)(ws + OFF_KQV);
  __bf16* qkvt  = (__bf16*)(ws + OFF_QKVT);
  float* rel    = (float*)(ws + OFF_REL);
  float* attn_t = (float*)(ws + OFF_ATTN);
  float* scale  = (float*)(ws + OFF_COEF);
  float* shift  = scale + 512;
  float* lsum   = (float*)(ws + OFF_LSUM);
  float* lsq    = (float*)(ws + OFF_LSQ);
  float* la     = (float*)(ws + OFF_LA);
  float* lc     = (float*)(ws + OFF_LC);
  float* out    = (float*)d_out;

  hipLaunchKernelGGL(k_kqv_gemm, dim3(256, 8), dim3(256), 0, stream, x, wk, kqv);
  hipLaunchKernelGGL(k_kqv_stats, dim3(512), dim3(256), 0, stream, kqv, kg, kb, scale, shift);
  hipLaunchKernelGGL(k_transpose_norm, dim3(1024, 16), dim3(256), 0, stream, kqv, scale, shift, qkvt);
  hipLaunchKernelGGL(k_rel_gather, dim3(64, 128), dim3(128), 0, stream, re, rel);
  hipLaunchKernelGGL(k_zero, dim3(1), dim3(64), 0, stream, lsum, 48);
  hipLaunchKernelGGL(k_logit_stats, dim3(256, 8), dim3(256), 0, stream, qkvt, rel, lsum, lsq);
  hipLaunchKernelGGL(k_logit_fin, dim3(1), dim3(32), 0, stream, lsum, lsq, lg, lb, la, lc);
  hipLaunchKernelGGL(k_attn, dim3(256, 8), dim3(256), 0, stream, qkvt, rel, la, lc, attn_t);
  hipLaunchKernelGGL(k_out_transpose, dim3(512, 16), dim3(256), 0, stream, attn_t, out);
}
